// ConvFrameMaskDecoder_62886911148230
// MI455X (gfx1250) — matrix operand order repaired and checked by reading, not compiled, not run on hardware
//
#include <hip/hip_runtime.h>
#include <math.h>

typedef __attribute__((ext_vector_type(16))) _Float16 v16h;
typedef __attribute__((ext_vector_type(8)))  _Float16 v8h;
typedef __attribute__((ext_vector_type(2)))  _Float16 v2h;
typedef __attribute__((ext_vector_type(8)))  float    v8f;

#define EPSV 1e-5f
#define BB 32
#define TT 48
#define LL 64
#define DHIDC 512
#define DEMBC 128
#define MCC 4
#define NBC 52
#define GDC 3
#define ACTO 322           // 4 + 6*52 + 2*3
#define ACTOP 384          // padded to 64
#define KIN 1152           // vis(512)+weighted(512)+e(128)
#define KCAT 1664          // inp(1152)+h(512)
#define MPIX 75264         // B*T*49
#define BT 1536            // B*T
#define FCK 3136           // 64*49

// Fragment-ready packed-B layout: tile = 16(n) x 32(k), k contiguous per n.
__device__ __forceinline__ int bpk_idx(int n, int k, int Npad) {
    return (((k >> 5) * (Npad >> 4) + (n >> 4)) << 9) | ((n & 15) << 5) | (k & 31);
}

// ---------------- output store functors ----------------
struct StoreF16RM { _Float16* c; int ld;
    __device__ void operator()(int r, int n, float v) const { c[(size_t)r * ld + n] = (_Float16)v; } };
struct StoreAct2 {              // [bt][n*49+p] NCHW-flatten for the FC input
    _Float16* c;
    __device__ void operator()(int r, int n, float v) const {
        int bt = r / 49, p = r - bt * 49;
        c[(size_t)bt * FCK + n * 49 + p] = (_Float16)v;
    } };
struct StoreF32RM { float* c; int ld;
    __device__ void operator()(int r, int n, float v) const { c[(size_t)r * ld + n] = v; } };
struct StoreActions {           // d_out actions [b][t][n]
    float* c; int t;
    __device__ void operator()(int r, int n, float v) const { c[((size_t)r * TT + t) * ACTO + n] = v; } };

// ---------------- fragment helpers ----------------
__device__ __forceinline__ v16h load_bfrag(const _Float16* __restrict__ Bp, int k0, int gn,
                                           int ntiles, int l16, int lhalf) {
    const _Float16* p = Bp + ((((size_t)(k0 >> 5) * ntiles) + (gn >> 4)) << 9)
                           + (l16 << 5) + (lhalf << 3);
    return make_afrag(*(const v8h*)p, *(const v8h*)(p + 16));     // two 16B aligned halves, table k order
}
__device__ __forceinline__ v16h make_afrag(v8h lo, v8h hi) {
    v16h a;
#pragma unroll
    for (int j = 0; j < 8; ++j) { a[j] = lo[j]; a[8 + j] = hi[j]; }
    return a;
}

// ================= GEMM variant 1: frames A (strided gather), double-buffered LDS ==
// Exact multiples required: M%64==0, N%64==0, K%32==0. bias mandatory.
template <class CS, bool RELU>
__global__ __launch_bounds__(128) void gemm_lds_frames(const float* __restrict__ frames,
                                                       const _Float16* __restrict__ Bp, int Npad,
                                                       const float* __restrict__ bias,
                                                       int K, CS cstore) {
    __shared__ _Float16 As[2][64][40];
    const int tid = threadIdx.x;
    const int wave = tid >> 5, lane = tid & 31;
    const int wm = (wave >> 1) * 32, wn = (wave & 1) * 32;
    const int bm = blockIdx.y * 64, bn = blockIdx.x * 64;
    const int lhalf = lane >> 4, l16 = lane & 15;
    const int ntiles = Npad >> 4;
    const int kb = lhalf * 8;

    float bv[2];
#pragma unroll
    for (int in = 0; in < 2; ++in) bv[in] = bias[bn + wn + in * 16 + l16];

    v8f acc[2][2];
#pragma unroll
    for (int i = 0; i < 2; ++i)
#pragma unroll
        for (int j = 0; j < 2; ++j)
#pragma unroll
            for (int r = 0; r < 8; ++r) acc[i][j][r] = 0.0f;

    // prologue: stage buffer 0 (k=0..31), m-fast (coalesced over pixels)
#pragma unroll
    for (int i = 0; i < 8; ++i) {
        int e = i * 128 + tid;
        int m = e & 63, kp = e >> 6;
        int row = bm + m;
        int bt = row / 49, p = row - bt * 49;
        const float* fp = frames + (size_t)(bt * 512 + 2 * kp) * 49 + p;
        v2h pk; pk[0] = (_Float16)fp[0]; pk[1] = (_Float16)fp[49];
        *(v2h*)&As[0][m][2 * kp] = pk;
    }
    v16h bf[2];
#pragma unroll
    for (int in = 0; in < 2; ++in) bf[in] = load_bfrag(Bp, 0, bn + wn + in * 16, ntiles, l16, lhalf);
    __syncthreads();

    int cur = 0;
    for (int k0 = 0; k0 < K; k0 += 32) {
        const int kn = k0 + 32;
        const bool has_next = kn < K;
        float nf0[8], nf1[8];
        v16h nbf[2];
        if (has_next) {
#pragma unroll
            for (int i = 0; i < 8; ++i) {
                int e = i * 128 + tid;
                int m = e & 63, kp = e >> 6;
                int row = bm + m;
                int bt = row / 49, p = row - bt * 49;
                const float* fp = frames + (size_t)(bt * 512 + kn + 2 * kp) * 49 + p;
                nf0[i] = fp[0]; nf1[i] = fp[49];
            }
#pragma unroll
            for (int in = 0; in < 2; ++in)
                nbf[in] = load_bfrag(Bp, kn, bn + wn + in * 16, ntiles, l16, lhalf);
        }
        v16h af[2];
#pragma unroll
        for (int im = 0; im < 2; ++im) {
            int arow = wm + im * 16 + l16;
            af[im] = make_afrag(*(const v8h*)&As[cur][arow][kb], *(const v8h*)&As[cur][arow][16 + kb]);
        }
#pragma unroll
        for (int im = 0; im < 2; ++im)
#pragma unroll
            for (int in = 0; in < 2; ++in)
                acc[im][in] = __builtin_amdgcn_wmma_f32_16x16x32_f16(
                    false, af[im], false, bf[in], (short)0, acc[im][in], false, false);
        if (has_next) {
#pragma unroll
            for (int i = 0; i < 8; ++i) {
                int e = i * 128 + tid;
                int m = e & 63, kp = e >> 6;
                v2h pk; pk[0] = (_Float16)nf0[i]; pk[1] = (_Float16)nf1[i];
                *(v2h*)&As[cur ^ 1][m][2 * kp] = pk;
            }
#pragma unroll
            for (int in = 0; in < 2; ++in) bf[in] = nbf[in];
        }
        cur ^= 1;
        __syncthreads();
    }
#pragma unroll
    for (int im = 0; im < 2; ++im)
#pragma unroll
        for (int in = 0; in < 2; ++in)
#pragma unroll
            for (int r = 0; r < 8; ++r) {
                int row = bm + wm + im * 16 + lhalf * 8 + r;
                int col = bn + wn + in * 16 + l16;
                float v = acc[im][in][r] + bv[in];
                if (RELU) v = v > 0.0f ? v : 0.0f;
                cstore(row, col, v);
            }
}

// ================= GEMM variant 2: f16 row-major A, double-buffered LDS ===========
template <class CS, bool RELU>
__global__ __launch_bounds__(128) void gemm_lds_f16(const _Float16* __restrict__ A, int lda,
                                                    const _Float16* __restrict__ Bp, int Npad,
                                                    const float* __restrict__ bias,
                                                    int K, CS cstore) {
    __shared__ _Float16 As[2][64][40];
    const int tid = threadIdx.x;
    const int wave = tid >> 5, lane = tid & 31;
    const int wm = (wave >> 1) * 32, wn = (wave & 1) * 32;
    const int bm = blockIdx.y * 64, bn = blockIdx.x * 64;
    const int lhalf = lane >> 4, l16 = lane & 15;
    const int ntiles = Npad >> 4;
    const int sub = (tid & 3) * 8, mr = tid >> 2;   // 4 threads/row, 8 f16 each
    const int kb = lhalf * 8;

    float bv[2];
#pragma unroll
    for (int in = 0; in < 2; ++in) bv[in] = bias[bn + wn + in * 16 + l16];

    v8f acc[2][2];
#pragma unroll
    for (int i = 0; i < 2; ++i)
#pragma unroll
        for (int j = 0; j < 2; ++j)
#pragma unroll
            for (int r = 0; r < 8; ++r) acc[i][j][r] = 0.0f;

#pragma unroll
    for (int i = 0; i < 2; ++i)
        *(v8h*)&As[0][mr + i * 32][sub] = *(const v8h*)(A + (size_t)(bm + mr + i * 32) * lda + sub);
    v16h bf[2];
#pragma unroll
    for (int in = 0; in < 2; ++in) bf[in] = load_bfrag(Bp, 0, bn + wn + in * 16, ntiles, l16, lhalf);
    __syncthreads();

    int cur = 0;
    for (int k0 = 0; k0 < K; k0 += 32) {
        const int kn = k0 + 32;
        const bool has_next = kn < K;
        v8h nA[2];
        v16h nbf[2];
        if (has_next) {
#pragma unroll
            for (int i = 0; i < 2; ++i)
                nA[i] = *(const v8h*)(A + (size_t)(bm + mr + i * 32) * lda + kn + sub);
#pragma unroll
            for (int in = 0; in < 2; ++in)
                nbf[in] = load_bfrag(Bp, kn, bn + wn + in * 16, ntiles, l16, lhalf);
        }
        v16h af[2];
#pragma unroll
        for (int im = 0; im < 2; ++im) {
            int arow = wm + im * 16 + l16;
            af[im] = make_afrag(*(const v8h*)&As[cur][arow][kb], *(const v8h*)&As[cur][arow][16 + kb]);
        }
#pragma unroll
        for (int im = 0; im < 2; ++im)
#pragma unroll
            for (int in = 0; in < 2; ++in)
                acc[im][in] = __builtin_amdgcn_wmma_f32_16x16x32_f16(
                    false, af[im], false, bf[in], (short)0, acc[im][in], false, false);
        if (has_next) {
#pragma unroll
            for (int i = 0; i < 2; ++i)
                *(v8h*)&As[cur ^ 1][mr + i * 32][sub] = nA[i];
#pragma unroll
            for (int in = 0; in < 2; ++in) bf[in] = nbf[in];
        }
        cur ^= 1;
        __syncthreads();
    }
#pragma unroll
    for (int im = 0; im < 2; ++im)
#pragma unroll
        for (int in = 0; in < 2; ++in)
#pragma unroll
            for (int r = 0; r < 8; ++r) {
                int row = bm + wm + im * 16 + lhalf * 8 + r;
                int col = bn + wn + in * 16 + l16;
                float v = acc[im][in][r] + bv[in];
                if (RELU) v = v > 0.0f ? v : 0.0f;
                cstore(row, col, v);
            }
}

// ================= GEMM variant 3: M=32, no LDS, register-pipelined ===============
template <class CS>
__global__ __launch_bounds__(128) void gemm_small(const _Float16* __restrict__ A, int lda,
                                                  const _Float16* __restrict__ Bp, int Npad,
                                                  const float* __restrict__ bias,
                                                  int N, int K, CS cstore) {
    const int tid = threadIdx.x;
    const int wave = tid >> 5, lane = tid & 31;
    const int lhalf = lane >> 4, l16 = lane & 15;
    const int gn = blockIdx.x * 64 + wave * 16;
    const int ntiles = Npad >> 4;
    const int col = gn + l16;
    const float bv = (col < N) ? bias[col] : 0.0f;

    v8f acc[2];
#pragma unroll
    for (int im = 0; im < 2; ++im)
#pragma unroll
        for (int r = 0; r < 8; ++r) acc[im][r] = 0.0f;

    v16h af[2], bf;
#pragma unroll
    for (int im = 0; im < 2; ++im) {
        const _Float16* ap = A + (size_t)(im * 16 + l16) * lda + lhalf * 8;
        af[im] = make_afrag(*(const v8h*)ap, *(const v8h*)(ap + 16));
    }
    bf = load_bfrag(Bp, 0, gn, ntiles, l16, lhalf);

    for (int k0 = 32; k0 < K; k0 += 32) {
        v16h naf[2], nbf;
#pragma unroll
        for (int im = 0; im < 2; ++im) {
            const _Float16* ap = A + (size_t)(im * 16 + l16) * lda + k0 + lhalf * 8;
            naf[im] = make_afrag(*(const v8h*)ap, *(const v8h*)(ap + 16));
        }
        nbf = load_bfrag(Bp, k0, gn, ntiles, l16, lhalf);
#pragma unroll
        for (int im = 0; im < 2; ++im)
            acc[im] = __builtin_amdgcn_wmma_f32_16x16x32_f16(
                false, af[im], false, bf, (short)0, acc[im], false, false);
        af[0] = naf[0]; af[1] = naf[1]; bf = nbf;
    }
#pragma unroll
    for (int im = 0; im < 2; ++im)
        acc[im] = __builtin_amdgcn_wmma_f32_16x16x32_f16(
            false, af[im], false, bf, (short)0, acc[im], false, false);

#pragma unroll
    for (int im = 0; im < 2; ++im)
#pragma unroll
        for (int r = 0; r < 8; ++r) {
            int row = im * 16 + lhalf * 8 + r;
            if (col < N) cstore(row, col, acc[im][r] + bv);
        }
}

// ---------------- weight prep (all write fragment-ready packed B) ----------------
__global__ void k_fold_conv(const float* __restrict__ w, const float* __restrict__ cb,
                            const float* __restrict__ g, const float* __restrict__ bb,
                            const float* __restrict__ m, const float* __restrict__ v,
                            _Float16* __restrict__ wo, float* __restrict__ bo, int K, int N) {
    int idx = blockIdx.x * blockDim.x + threadIdx.x;
    if (idx < K * N) {
        int k = idx / N, n = idx - k * N;
        float s = g[n] * rsqrtf(v[n] + EPSV);
        wo[bpk_idx(n, k, N)] = (_Float16)(w[n * K + k] * s);
    }
    if (idx < N) {
        float s = g[idx] * rsqrtf(v[idx] + EPSV);
        bo[idx] = (cb[idx] - m[idx]) * s + bb[idx];
    }
}
__global__ void k_pack_w(const float* __restrict__ w, _Float16* __restrict__ o,
                         int N, int Npad, int K) {   // w row-major [N,K]; zero-pad to Npad
    int idx = blockIdx.x * blockDim.x + threadIdx.x;
    if (idx < Npad * K) {
        int k = idx / Npad, n = idx - k * Npad;
        float v = (n < N) ? w[n * K + k] : 0.0f;
        o[bpk_idx(n, k, Npad)] = (_Float16)v;
    }
}
__global__ void k_pack_wcat(const float* __restrict__ wih, const float* __restrict__ whh,
                            _Float16* __restrict__ o) {
    int idx = blockIdx.x * blockDim.x + threadIdx.x;
    if (idx < KCAT * 2048) {
        int k = idx / 2048, n = idx - k * 2048;
        float v = (k < KIN) ? wih[n * KIN + k] : whh[n * DHIDC + (k - KIN)];
        o[bpk_idx(n, k, 2048)] = (_Float16)v;
    }
}
__global__ void k_add_bias(const float* a, const float* b, float* o, int n) {
    int i = blockIdx.x * blockDim.x + threadIdx.x;
    if (i < n) o[i] = a[i] + b[i];
}
__global__ void k_init(const float* h0, const float* c0, const float* go,
                       float* h, float* c, float* e) {
    int i = blockIdx.x * blockDim.x + threadIdx.x;
    if (i < BB * DHIDC) { h[i] = h0[i]; c[i] = c0[i]; }
    if (i < BB * DEMBC) e[i] = go[i & 127];
}

// ---------------- per-step: attention + build x = [vis, weighted, e, h] ----------------
__global__ __launch_bounds__(256) void k_attn(const float* __restrict__ enc,
                                              const float* __restrict__ hfc_w,
                                              const float* __restrict__ hfc_b,
                                              const float* __restrict__ h,
                                              const float* __restrict__ vis,
                                              const float* __restrict__ e,
                                              _Float16* __restrict__ x,
                                              float* __restrict__ scores, int t) {
    int b = blockIdx.x, tid = threadIdx.x;
    __shared__ float sh_h[DHIDC], sh_hq[DHIDC], sc[LL], red[2];
    for (int d = tid; d < DHIDC; d += 256) sh_h[d] = h[b * DHIDC + d];
    __syncthreads();
    for (int j = tid; j < DHIDC; j += 256) {
        const float* wr = hfc_w + j * DHIDC;
        float s = hfc_b[j];
        for (int k = 0; k < DHIDC; ++k) s += sh_h[k] * wr[k];
        sh_hq[j] = s;
    }
    __syncthreads();
    if (tid < LL) {
        const float* er = enc + (b * LL + tid) * DHIDC;
        float s = 0.f;
        for (int k = 0; k < DHIDC; ++k) s += er[k] * sh_hq[k];
        sc[tid] = s;
    }
    __syncthreads();
    if (tid == 0) { float mx = sc[0]; for (int l = 1; l < LL; ++l) mx = fmaxf(mx, sc[l]); red[0] = mx; }
    __syncthreads();
    if (tid < LL) sc[tid] = expf(sc[tid] - red[0]);
    __syncthreads();
    if (tid == 0) { float s = 0.f; for (int l = 0; l < LL; ++l) s += sc[l]; red[1] = s; }
    __syncthreads();
    if (tid < LL) { sc[tid] /= red[1]; scores[(b * TT + t) * LL + tid] = sc[tid]; }
    __syncthreads();
    _Float16* xb = x + b * KCAT;
    for (int d = tid; d < DHIDC; d += 256) {
        float w = 0.f;
        for (int l = 0; l < LL; ++l) w += sc[l] * enc[(b * LL + l) * DHIDC + d];
        xb[512 + d]  = (_Float16)w;
        xb[d]        = (_Float16)vis[(b * TT + t) * DHIDC + d];
        xb[1152 + d] = (_Float16)sh_h[d];
    }
    if (tid < DEMBC) xb[1024 + tid] = (_Float16)e[b * DEMBC + tid];
}

// ---------------- per-step: LSTM pointwise + build cont = [h_new, inp] ----------------
__global__ void k_lstm(const float* __restrict__ gates, const _Float16* __restrict__ x,
                       float* h, float* c, _Float16* cont) {
    int idx = blockIdx.x * blockDim.x + threadIdx.x;
    if (idx >= BB * KCAT) return;
    int b = idx / KCAT, d = idx - b * KCAT;
    if (d < DHIDC) {
        const float* g = gates + b * 2048;
        float gi = 1.f / (1.f + expf(-g[d]));
        float gf = 1.f / (1.f + expf(-g[512 + d]));
        float gg = tanhf(g[1024 + d]);
        float go = 1.f / (1.f + expf(-g[1536 + d]));
        float cn = gf * c[b * DHIDC + d] + gi * gg;
        float hn = go * tanhf(cn);
        c[b * DHIDC + d] = cn;
        h[b * DHIDC + d] = hn;
        cont[b * KCAT + d] = (_Float16)hn;
    } else {
        cont[b * KCAT + d] = x[b * KCAT + (d - 512)];
    }
}

// ---------------- per-step: greedy decode -> embedding gather ----------------
__global__ __launch_bounds__(128) void k_decode(const float* __restrict__ actions, int t,
                                                const float* __restrict__ em_mode,
                                                const float* __restrict__ em_xy,
                                                const float* __restrict__ em_yaw,
                                                const float* __restrict__ em_eff,
                                                const float* __restrict__ em_gd,
                                                const float* __restrict__ em_ud,
                                                _Float16* __restrict__ embcat) {
    int b = blockIdx.x, tid = threadIdx.x;
    const float* a = actions + ((size_t)b * TT + t) * ACTO;
    __shared__ int idxs[9];
    if (tid == 0) {
        int bi = 0; float bv = a[0];
        for (int i = 1; i < MCC; ++i) if (a[i] > bv) { bv = a[i]; bi = i; }
        idxs[0] = bi;
        for (int jb = 0; jb < 6; ++jb) {
            const float* p = a + MCC + jb * NBC;
            bi = 0; bv = p[0];
            for (int i = 1; i < NBC; ++i) if (p[i] > bv) { bv = p[i]; bi = i; }
            idxs[1 + jb] = bi;
        }
        for (int jg = 0; jg < 2; ++jg) {
            const float* p = a + MCC + 6 * NBC + jg * GDC;
            bi = 0; bv = p[0];
            for (int i = 1; i < GDC; ++i) if (p[i] > bv) { bv = p[i]; bi = i; }
            idxs[7 + jg] = bi;
        }
    }
    __syncthreads();
    for (int i = tid; i < 9 * DEMBC; i += 128) {
        int slot = i >> 7, d = i & 127;
        const float* tab; int id;
        if (slot == 0)      { tab = em_mode; id = idxs[0]; }
        else if (slot <= 2) { tab = em_xy;   id = idxs[slot]; }
        else if (slot == 3) { tab = em_yaw;  id = idxs[3]; }
        else if (slot <= 6) { tab = em_eff;  id = idxs[slot]; }
        else if (slot == 7) { tab = em_gd;   id = idxs[7]; }
        else                { tab = em_ud;   id = idxs[8]; }
        embcat[b * (9 * DEMBC) + i] = (_Float16)tab[id * DEMBC + d];
    }
}

__global__ void k_final(const float* h, const float* c, float* oh, float* oc) {
    int i = blockIdx.x * blockDim.x + threadIdx.x;
    if (i < BB * DHIDC) { oh[i] = h[i]; oc[i] = c[i]; }
}

// ---------------- host launcher ----------------
extern "C" void kernel_launch(void* const* d_in, const int* in_sizes, int n_in,
                              void* d_out, int out_size, void* d_ws, size_t ws_size,
                              hipStream_t stream) {
    const float* enc     = (const float*)d_in[0];
    const float* frames  = (const float*)d_in[1];
    const float* h0      = (const float*)d_in[3];
    const float* c0      = (const float*)d_in[4];
    const float* conv1_w = (const float*)d_in[5];
    const float* conv1_b = (const float*)d_in[6];
    const float* bn1_g   = (const float*)d_in[7];
    const float* bn1_b   = (const float*)d_in[8];
    const float* bn1_m   = (const float*)d_in[9];
    const float* bn1_v   = (const float*)d_in[10];
    const float* conv2_w = (const float*)d_in[11];
    const float* conv2_b = (const float*)d_in[12];
    const float* bn2_g   = (const float*)d_in[13];
    const float* bn2_b   = (const float*)d_in[14];
    const float* bn2_m   = (const float*)d_in[15];
    const float* bn2_v   = (const float*)d_in[16];
    const float* fc_w    = (const float*)d_in[17];
    const float* fc_b    = (const float*)d_in[18];
    const float* hfc_w   = (const float*)d_in[19];
    const float* hfc_b   = (const float*)d_in[20];
    const float* w_ih    = (const float*)d_in[21];
    const float* w_hh    = (const float*)d_in[22];
    const float* b_ih    = (const float*)d_in[23];
    const float* b_hh    = (const float*)d_in[24];
    const float* actor_w = (const float*)d_in[25];
    const float* actor_b = (const float*)d_in[26];
    const float* adpt_w  = (const float*)d_in[27];
    const float* adpt_b  = (const float*)d_in[28];
    const float* go      = (const float*)d_in[29];
    const float* em_mode = (const float*)d_in[30];
    const float* em_xy   = (const float*)d_in[31];
    const float* em_yaw  = (const float*)d_in[32];
    const float* em_eff  = (const float*)d_in[33];
    const float* em_gd   = (const float*)d_in[34];
    const float* em_ud   = (const float*)d_in[35];

    float* out = (float*)d_out;
    float* out_actions = out;                       // [32,48,322]
    float* out_scores  = out + BB * TT * ACTO;      // [32,48,64]
    float* out_h       = out_scores + BB * TT * LL; // [32,512]
    float* out_c       = out_h + BB * DHIDC;        // [32,512]

    char* ws = (char*)d_ws;
    size_t off = 0;
    auto alloc = [&](size_t bytes) -> char* {
        char* p = ws + off;
        off = (off + bytes + 255) & ~(size_t)255;
        return p;
    };
    _Float16* w1     = (_Float16*)alloc(512 * 256 * 2);          // packed
    float*    b1     = (float*)alloc(256 * 4);
    _Float16* w2     = (_Float16*)alloc(256 * 64 * 2);           // packed
    float*    b2     = (float*)alloc(64 * 4);
    _Float16* fcw    = (_Float16*)alloc((size_t)FCK * 512 * 2);  // packed
    _Float16* wcat   = (_Float16*)alloc((size_t)KCAT * 2048 * 2);// packed
    float*    bg     = (float*)alloc(2048 * 4);
    _Float16* actw   = (_Float16*)alloc((size_t)KCAT * ACTOP * 2); // packed, padded
    _Float16* adpw   = (_Float16*)alloc((size_t)KIN * DEMBC * 2);  // packed
    _Float16* act1   = (_Float16*)alloc((size_t)MPIX * 256 * 2);
    _Float16* act2   = (_Float16*)alloc((size_t)BT * FCK * 2);
    float*    vis    = (float*)alloc((size_t)BT * DHIDC * 4);
    float*    hbuf   = (float*)alloc(BB * DHIDC * 4);
    float*    cbuf   = (float*)alloc(BB * DHIDC * 4);
    float*    ebuf   = (float*)alloc(BB * DEMBC * 4);
    _Float16* x16    = (_Float16*)alloc(BB * KCAT * 2);
    _Float16* cont16 = (_Float16*)alloc(BB * KCAT * 2);
    float*    gates  = (float*)alloc(BB * 2048 * 4);
    _Float16* embcat = (_Float16*)alloc(BB * 9 * DEMBC * 2);
    (void)ws_size; (void)in_sizes; (void)n_in; (void)out_size;

    // ---- weight prep (fold BN, convert, pack fragment-ready) ----
    k_fold_conv<<<(512 * 256 + 255) / 256, 256, 0, stream>>>(conv1_w, conv1_b, bn1_g, bn1_b,
                                                             bn1_m, bn1_v, w1, b1, 512, 256);
    k_fold_conv<<<(256 * 64 + 255) / 256, 256, 0, stream>>>(conv2_w, conv2_b, bn2_g, bn2_b,
                                                            bn2_m, bn2_v, w2, b2, 256, 64);
    k_pack_w<<<(FCK * 512 + 255) / 256, 256, 0, stream>>>(fc_w, fcw, 512, 512, FCK);
    k_pack_wcat<<<(KCAT * 2048 + 255) / 256, 256, 0, stream>>>(w_ih, w_hh, wcat);
    k_add_bias<<<(2048 + 255) / 256, 256, 0, stream>>>(b_ih, b_hh, bg, 2048);
    k_pack_w<<<(KCAT * ACTOP + 255) / 256, 256, 0, stream>>>(actor_w, actw, ACTO, ACTOP, KCAT);
    k_pack_w<<<(KIN * DEMBC + 255) / 256, 256, 0, stream>>>(adpt_w, adpw, DEMBC, DEMBC, KIN);
    k_init<<<(BB * DHIDC + 255) / 256, 256, 0, stream>>>(h0, c0, go, hbuf, cbuf, ebuf);

    // ---- phase 1: visual encoder (batched over B*T, all dims exact multiples) ----
    gemm_lds_frames<StoreF16RM, true><<<dim3(256 / 64, MPIX / 64), 128, 0, stream>>>(
        frames, w1, 256, b1, 512, StoreF16RM{act1, 256});
    gemm_lds_f16<StoreAct2, true><<<dim3(1, MPIX / 64), 128, 0, stream>>>(
        act1, 256, w2, 64, b2, 256, StoreAct2{act2});
    gemm_lds_f16<StoreF32RM, false><<<dim3(512 / 64, BT / 64), 128, 0, stream>>>(
        act2, FCK, fcw, 512, fc_b, FCK, StoreF32RM{vis, 512});

    // ---- phase 2: recurrence over T ----
    for (int t = 0; t < TT; ++t) {
        k_attn<<<BB, 256, 0, stream>>>(enc, hfc_w, hfc_b, hbuf, vis, ebuf, x16, out_scores, t);
        gemm_small<StoreF32RM><<<dim3(2048 / 64, 1), 128, 0, stream>>>(
            x16, KCAT, wcat, 2048, bg, 2048, KCAT, StoreF32RM{gates, 2048});
        k_lstm<<<(BB * KCAT + 255) / 256, 256, 0, stream>>>(gates, x16, hbuf, cbuf, cont16);
        gemm_small<StoreActions><<<dim3(ACTOP / 64, 1), 128, 0, stream>>>(
            cont16, KCAT, actw, ACTOP, actor_b, ACTO, KCAT, StoreActions{out_actions, t});
        k_decode<<<BB, 128, 0, stream>>>(out_actions, t, em_mode, em_xy, em_yaw, em_eff,
                                         em_gd, em_ud, embcat);
        gemm_small<StoreF32RM><<<dim3(DEMBC / 64, 1), 128, 0, stream>>>(
            embcat, KIN, adpw, DEMBC, adpt_b, DEMBC, KIN, StoreF32RM{ebuf, DEMBC});
    }
    k_final<<<(BB * DHIDC + 255) / 256, 256, 0, stream>>>(hbuf, cbuf, out_h, out_c);
}
